// QuanvolutionHybridClassifier_65481071396430
// MI455X (gfx1250) — matrix operand order repaired and checked by reading, not compiled, not run on hardware
//
#include <hip/hip_runtime.h>
#include <hip/hip_bf16.h>

typedef __attribute__((ext_vector_type(16))) _Float16 v16h;
typedef __attribute__((ext_vector_type(8)))  _Float16 v8h;
typedef __attribute__((ext_vector_type(4)))  _Float16 v4h;
typedef __attribute__((ext_vector_type(8)))  float    v8f;

#define BATCH  4096
#define NPATCH 196
#define KREAL  784
#define KPAD   800
#define HID    512
#define NCLS   10

// ---------------------------------------------------------------------------
// 4-qubit statevector helpers (fully unrolled, state lives in 16 registers)
// wire w <-> bit (1 << (3-w))  (wire 0 is most-significant, per reference)
// ---------------------------------------------------------------------------
__device__ __forceinline__ void qc_ry(float* s, float c, float sn, int bit) {
#pragma unroll
  for (int i = 0; i < 16; ++i) {
    if (i & bit) continue;
    const int j = i | bit;
    const float s0 = s[i], s1 = s[j];
    s[i] = c * s0 - sn * s1;
    s[j] = sn * s0 + c * s1;
  }
}

__device__ __forceinline__ void qc_cnot(float* s, int cbit, int tbit) {
#pragma unroll
  for (int i = 0; i < 16; ++i) {
    if ((i & cbit) && !(i & tbit)) {
      const int j = i | tbit;
      const float t = s[i]; s[i] = s[j]; s[j] = t;
    }
  }
}

// ---------------------------------------------------------------------------
// Kernel 1: quanvolution features -> f16, row stride KPAD (pad filled later)
// ---------------------------------------------------------------------------
__global__ void k_quanv(const float* __restrict__ x,
                        const float* __restrict__ params,
                        _Float16* __restrict__ feat) {
  const int idx = blockIdx.x * blockDim.x + threadIdx.x;
  if (idx >= BATCH * NPATCH) return;
  const int b = idx / NPATCH;
  const int p = idx - b * NPATCH;
  const int pr = p / 14, pc = p - pr * 14;
  const float* xb = x + (size_t)b * 784;

  float ang[4];
  ang[0] = xb[(2 * pr)     * 28 + 2 * pc];
  ang[1] = xb[(2 * pr)     * 28 + 2 * pc + 1];
  ang[2] = xb[(2 * pr + 1) * 28 + 2 * pc];
  ang[3] = xb[(2 * pr + 1) * 28 + 2 * pc + 1];

  float s[16];
#pragma unroll
  for (int i = 0; i < 16; ++i) s[i] = 0.0f;
  s[0] = 1.0f;

  // data encoding
#pragma unroll
  for (int w = 0; w < 4; ++w) {
    float sn, c;
    __sincosf(0.5f * ang[w], &sn, &c);
    qc_ry(s, c, sn, 1 << (3 - w));
  }

  // variational layers
#pragma unroll
  for (int layer = 0; layer < 2; ++layer) {
#pragma unroll
    for (int w = 0; w < 4; ++w) {
      float sn, c;
      __sincosf(0.5f * params[layer * 4 + w], &sn, &c);
      qc_ry(s, c, sn, 1 << (3 - w));
    }
    qc_cnot(s, 1 << 3, 1 << 2);  // CNOT(0,1)
    qc_cnot(s, 1 << 2, 1 << 1);  // CNOT(1,2)
    qc_cnot(s, 1 << 1, 1 << 0);  // CNOT(2,3)
    qc_cnot(s, 1 << 0, 1 << 3);  // CNOT(3,0)
  }

  // <Z_w> expectations -> 4 f16 features, one 8B store
  v4h out;
#pragma unroll
  for (int w = 0; w < 4; ++w) {
    const int bit = 1 << (3 - w);
    float e = 0.0f;
#pragma unroll
    for (int i = 0; i < 16; ++i) {
      const float pr2 = s[i] * s[i];
      e += (i & bit) ? -pr2 : pr2;
    }
    out[w] = (_Float16)e;
  }
  *(v4h*)(feat + (size_t)b * KPAD + p * 4) = out;
}

// zero the 16 pad columns of each feature row
__global__ void k_padfeat(_Float16* __restrict__ feat) {
  const int idx = blockIdx.x * blockDim.x + threadIdx.x;
  if (idx >= BATCH * (KPAD - KREAL)) return;
  const int b = idx >> 4;
  const int k = idx & 15;
  feat[(size_t)b * KPAD + KREAL + k] = (_Float16)0.0f;
}

// convert fc1_w (512x784 f32) -> f16 with K padded to 800 (zeros)
__global__ void k_cvtw(const float* __restrict__ w1, _Float16* __restrict__ w1h) {
  const int idx = blockIdx.x * blockDim.x + threadIdx.x;
  if (idx >= HID * KPAD) return;
  const int n = idx / KPAD;
  const int k = idx - n * KPAD;
  w1h[idx] = (k < KREAL) ? (_Float16)w1[(size_t)n * KREAL + k] : (_Float16)0.0f;
}

// ---------------------------------------------------------------------------
// Kernel 2: h = relu(feat @ W1^T + b1) via v_wmma_f32_16x16x32_f16
// one wave per 16(M) x 64(N) tile; A fragment reused across 4 N-tiles
// ---------------------------------------------------------------------------
__global__ void __launch_bounds__(32) k_gemm1(const _Float16* __restrict__ feat,
                                              const _Float16* __restrict__ w1h,
                                              const float* __restrict__ bias1,
                                              float* __restrict__ h) {
  const int lane = threadIdx.x;       // 0..31 (wave32)
  const int l    = lane & 15;
  const int hi   = lane >> 4;
  const int m0   = blockIdx.x * 16;   // batch tile
  const int n0   = blockIdx.y * 64;   // hidden tile group

  v8f acc[4] = {};
  const _Float16* arow = feat + (size_t)(m0 + l) * KPAD;

  for (int k0 = 0; k0 < KPAD; k0 += 32) {
    // A fragment (16x32 f16): elems 0-15 = K k0+16*hi.. (contiguous)
    const v16h a = *(const v16h*)(arow + k0 + 16 * hi);

#pragma unroll
    for (int t = 0; t < 4; ++t) {
      // B fragment (32x16 f16): lane holds column n, 16 contiguous K values
      const int col = n0 + t * 16 + l;
      const v16h bfrag = *(const v16h*)(w1h + (size_t)col * KPAD + k0 + 16 * hi);
      acc[t] = __builtin_amdgcn_wmma_f32_16x16x32_f16(
          /*neg_a=*/false, a, /*neg_b=*/false, bfrag,
          /*c_mod=*/(short)0, acc[t], /*reuse_a=*/false, /*reuse_b=*/false);
    }
  }

  // epilogue: bias + ReLU, scatter per C/D layout (vgpr r -> row m0 + r + 8*hi)
#pragma unroll
  for (int t = 0; t < 4; ++t) {
    const int col = n0 + t * 16 + l;
    const float bv = bias1[col];
#pragma unroll
    for (int r = 0; r < 8; ++r) {
      float v = acc[t][r] + bv;
      v = v > 0.0f ? v : 0.0f;
      h[(size_t)(m0 + r + 8 * hi) * HID + col] = v;
    }
  }
}

// ---------------------------------------------------------------------------
// Kernel 3: logits = h @ W2^T + b2 ; out = log_softmax(logits)
// one wave per batch row; butterfly shuffle reduction
// ---------------------------------------------------------------------------
__global__ void __launch_bounds__(32) k_head(const float* __restrict__ h,
                                             const float* __restrict__ w2,
                                             const float* __restrict__ b2,
                                             float* __restrict__ out) {
  const int b = blockIdx.x;
  const int lane = threadIdx.x;
  const float* hr = h + (size_t)b * HID;

  float acc[NCLS];
#pragma unroll
  for (int j = 0; j < NCLS; ++j) acc[j] = 0.0f;

  for (int k = lane; k < HID; k += 32) {
    const float hv = hr[k];
#pragma unroll
    for (int j = 0; j < NCLS; ++j) acc[j] += hv * w2[(size_t)j * HID + k];
  }

#pragma unroll
  for (int j = 0; j < NCLS; ++j) {
#pragma unroll
    for (int off = 16; off > 0; off >>= 1)
      acc[j] += __shfl_xor(acc[j], off, 32);
  }

  float logits[NCLS];
  float m = -1e30f;
#pragma unroll
  for (int j = 0; j < NCLS; ++j) {
    logits[j] = acc[j] + b2[j];
    m = fmaxf(m, logits[j]);
  }
  float sum = 0.0f;
#pragma unroll
  for (int j = 0; j < NCLS; ++j) sum += __expf(logits[j] - m);
  const float lse = __logf(sum) + m;

  if (lane < NCLS) out[(size_t)b * NCLS + lane] = logits[lane] - lse;
}

// ---------------------------------------------------------------------------
extern "C" void kernel_launch(void* const* d_in, const int* in_sizes, int n_in,
                              void* d_out, int out_size, void* d_ws, size_t ws_size,
                              hipStream_t stream) {
  const float* x      = (const float*)d_in[0];
  const float* params = (const float*)d_in[1];
  const float* fc1_w  = (const float*)d_in[2];
  const float* fc1_b  = (const float*)d_in[3];
  const float* fc2_w  = (const float*)d_in[4];
  const float* fc2_b  = (const float*)d_in[5];
  float* out = (float*)d_out;

  _Float16* feat = (_Float16*)d_ws;                       // 4096*800 f16
  _Float16* w1h  = feat + (size_t)BATCH * KPAD;           //  512*800 f16
  float*    h    = (float*)(w1h + (size_t)HID * KPAD);    // 4096*512 f32

  k_quanv<<<(BATCH * NPATCH + 255) / 256, 256, 0, stream>>>(x, params, feat);
  k_padfeat<<<(BATCH * (KPAD - KREAL) + 255) / 256, 256, 0, stream>>>(feat);
  k_cvtw<<<(HID * KPAD + 255) / 256, 256, 0, stream>>>(fc1_w, w1h);

  dim3 g1(BATCH / 16, HID / 64);
  k_gemm1<<<g1, 32, 0, stream>>>(feat, w1h, fc1_b, h);

  k_head<<<BATCH, 32, 0, stream>>>(h, fc2_w, fc2_b, out);
}
